// MultiHeadAttention_12713103196391
// MI455X (gfx1250) — compile-verified, matrix operand order checked by reading, not run on hardware
//
#include <hip/hip_runtime.h>
#include <hip/hip_bf16.h>

// MHA forward for B=4, H=16, S=2048, D=1024, D_K=64 on gfx1250 (MI455X).
// Strategy: f16 WMMA for all GEMMs; fused scores+softmax+PV with the 16x2048
// probability strip resident in LDS (128KB of the 320KB/WGP). p_attn (1.07 GB,
// the HBM-traffic floor of this problem) is written exactly once, via
// GLOBAL_STORE_ASYNC_FROM_LDS_B128 so the store drains concurrently with the
// PV WMMA stage instead of serializing in front of it.

#define BATCH 4
#define HEADS 16
#define SEQ   2048
#define DMODEL 1024
#define DK    64

typedef __attribute__((ext_vector_type(16))) _Float16 v16h;
typedef __attribute__((ext_vector_type(8)))  _Float16 v8h;
typedef __attribute__((ext_vector_type(8)))  float    v8f;

// ---------------------------------------------------------------------------
// WMMA wrapper: D = A(16x32 f16) * B(32x16 f16) + C(16x16 f32)
// ---------------------------------------------------------------------------
__device__ __forceinline__ v8f wmma_f16(v16h a, v16h b, v8f c) {
  return __builtin_amdgcn_wmma_f32_16x16x32_f16(
      /*neg_a=*/false, a, /*neg_b=*/false, b,
      /*c_mod=*/(short)0, c, /*reuse_a=*/false, /*reuse_b=*/false);
}

// ---------------------------------------------------------------------------
// Fragment loaders (wave32). ISA 16-bit A-matrix 16x32 layout:
//   lane = g*16 + r  (r = row M, g = K-group)
//   elems 0..7  -> K = g*8 + i ; elems 8..15 -> K = 16 + g*8 + i
// "NT" loader: element (row, k) read from base[row*ld + k] (k contiguous).
// ---------------------------------------------------------------------------
__device__ __forceinline__ v16h frag_nt_f16(const _Float16* base, int ld) {
  const int lane = threadIdx.x & 31;
  const int r = lane & 15, g = lane >> 4;
  const _Float16* row = base + (size_t)r * ld;
  v8h lo = *(const v8h*)(row + g * 8);
  v8h hi = *(const v8h*)(row + 16 + g * 8);
  v16h f;
#pragma unroll
  for (int i = 0; i < 8; ++i) { f[i] = lo[i]; f[8 + i] = hi[i]; }
  return f;
}

// Same layout, but source is f32 (converted to f16 in registers).
__device__ __forceinline__ v16h frag_nt_f32(const float* base, int ld) {
  const int lane = threadIdx.x & 31;
  const int r = lane & 15, g = lane >> 4;
  const float* row = base + (size_t)r * ld;
  float4 x0 = *(const float4*)(row + g * 8);
  float4 x1 = *(const float4*)(row + g * 8 + 4);
  float4 y0 = *(const float4*)(row + 16 + g * 8);
  float4 y1 = *(const float4*)(row + 16 + g * 8 + 4);
  v16h f;
  f[0]  = (_Float16)x0.x; f[1]  = (_Float16)x0.y;
  f[2]  = (_Float16)x0.z; f[3]  = (_Float16)x0.w;
  f[4]  = (_Float16)x1.x; f[5]  = (_Float16)x1.y;
  f[6]  = (_Float16)x1.z; f[7]  = (_Float16)x1.w;
  f[8]  = (_Float16)y0.x; f[9]  = (_Float16)y0.y;
  f[10] = (_Float16)y0.z; f[11] = (_Float16)y0.w;
  f[12] = (_Float16)y1.x; f[13] = (_Float16)y1.y;
  f[14] = (_Float16)y1.z; f[15] = (_Float16)y1.w;
  return f;
}

// B fragment for a row-major K x N source: element (k, n) at base[k*ld + n].
__device__ __forceinline__ v16h fragB_kn_f16(const _Float16* base, int ld) {
  const int lane = threadIdx.x & 31;
  const int n = lane & 15, g = lane >> 4;
  v16h f;
#pragma unroll
  for (int i = 0; i < 8; ++i)
    f[i] = base[(size_t)(g * 8 + i) * ld + n];
#pragma unroll
  for (int i = 0; i < 8; ++i)
    f[8 + i] = base[(size_t)(16 + g * 8 + i) * ld + n];
  return f;
}

// ---------------------------------------------------------------------------
// Kernel 1: fused QKV projection.  out = in @ W.T  (NT GEMM, 8192x1024x1024)
// z = 0/1/2 selects (query,Wq)->Qh, (key,Wk)->Kh, (value,Wv)->Vh.
// Output layout: head-major f16 [B, H, S, DK].
// ---------------------------------------------------------------------------
__global__ void proj_qkv(const float* __restrict__ q_in,
                         const float* __restrict__ k_in,
                         const float* __restrict__ v_in,
                         const float* __restrict__ w_in,
                         _Float16* __restrict__ qh,
                         _Float16* __restrict__ kh,
                         _Float16* __restrict__ vh) {
  const int z = blockIdx.z;
  const float* in = (z == 0) ? q_in : (z == 1) ? k_in : v_in;
  _Float16* out = (z == 0) ? qh : (z == 1) ? kh : vh;
  const float* wsel = w_in + (size_t)z * DMODEL * DMODEL;

  const int wave = threadIdx.x >> 5;
  const int lane = threadIdx.x & 31;
  const int m0 = blockIdx.y * 64 + wave * 16;      // token-row base
  const int nblk = blockIdx.x * 64;                // output-col base

  v8f acc[4] = {v8f{}, v8f{}, v8f{}, v8f{}};

  for (int k = 0; k < DMODEL; k += 32) {
    v16h a = frag_nt_f32(in + (size_t)m0 * DMODEL + k, DMODEL);
#pragma unroll
    for (int j = 0; j < 4; ++j) {
      v16h b = frag_nt_f32(wsel + (size_t)(nblk + j * 16) * DMODEL + k, DMODEL);
      acc[j] = wmma_f16(a, b, acc[j]);
    }
  }

  const int nl = lane & 15, g = lane >> 4;
#pragma unroll
  for (int j = 0; j < 4; ++j) {
#pragma unroll
    for (int r = 0; r < 8; ++r) {
      int m = m0 + r + 8 * g;
      int n = nblk + j * 16 + nl;
      int b_ = m >> 11, s = m & (SEQ - 1);
      int h = n >> 6, dk = n & (DK - 1);
      out[((((size_t)b_ * HEADS + h) * SEQ) + s) * DK + dk] =
          (_Float16)acc[j][r];
    }
  }
}

// ---------------------------------------------------------------------------
// Kernel 2: fused attention for one (b,h) x 16 query rows.
//   1) scores(16 x 2048) = scale * Q K^T   -> LDS (128 KB)
//   2) softmax rows in LDS
//   3) launch async LDS->HBM stores of the normalized strip (ASYNCcnt)
//   4) x(16 x 64) = P @ V via WMMA while the async stores drain
//   5) s_wait_asynccnt 0
// Block = 128 thr (4 waves), dynamic LDS = 16*2048*4 bytes.
// ---------------------------------------------------------------------------
__global__ void attn_fused(const _Float16* __restrict__ qh,
                           const _Float16* __restrict__ kh,
                           const _Float16* __restrict__ vh,
                           float* __restrict__ out_p,
                           float* __restrict__ out_x) {
  extern __shared__ float sm[];                 // [16][2048] score strip
  __shared__ float red[16][8];

  const int wave = threadIdx.x >> 5;
  const int lane = threadIdx.x & 31;
  const int bh = blockIdx.z;                    // b*HEADS + h
  const int m0 = blockIdx.y * 16;               // query-row base

  const _Float16* qbase = qh + ((size_t)bh * SEQ + m0) * DK;
  const _Float16* kbase = kh + (size_t)bh * SEQ * DK;
  const _Float16* vbase = vh + (size_t)bh * SEQ * DK;

  // ---- 1) QK^T into LDS ----
  v16h aq0 = frag_nt_f16(qbase, DK);            // K dims 0..31
  v16h aq1 = frag_nt_f16(qbase + 32, DK);       // K dims 32..63
  const int nl = lane & 15, g = lane >> 4;
  for (int n0 = wave * 16; n0 < SEQ; n0 += 64) {
    v16h b0 = frag_nt_f16(kbase + (size_t)n0 * DK, DK);
    v16h b1 = frag_nt_f16(kbase + (size_t)n0 * DK + 32, DK);
    v8f c = {};
    c = wmma_f16(aq0, b0, c);
    c = wmma_f16(aq1, b1, c);
#pragma unroll
    for (int r = 0; r < 8; ++r)
      sm[(r + 8 * g) * SEQ + n0 + nl] = c[r] * 0.125f;  // 1/sqrt(64)
  }
  __syncthreads();

  // ---- 2) softmax over each of the 16 LDS rows (8 threads / row) ----
  {
    const int row = threadIdx.x >> 3;   // 0..15
    const int sub = threadIdx.x & 7;    // 0..7
    float* r = sm + row * SEQ;

    float mx = -3.0e38f;
    for (int i = sub; i < SEQ; i += 8) mx = fmaxf(mx, r[i]);
    red[row][sub] = mx;
    __syncthreads();
    if (sub == 0) {
      float t = red[row][0];
#pragma unroll
      for (int i = 1; i < 8; ++i) t = fmaxf(t, red[row][i]);
      red[row][0] = t;
    }
    __syncthreads();
    const float m = red[row][0];

    float sum = 0.f;
    for (int i = sub; i < SEQ; i += 8) {
      float e = __expf(r[i] - m);
      r[i] = e;
      sum += e;
    }
    __syncthreads();
    red[row][sub] = sum;
    __syncthreads();
    if (sub == 0) {
      float t = 0.f;
#pragma unroll
      for (int i = 1; i < 8; ++i) t += red[row][i];
      red[row][0] += t;
    }
    __syncthreads();
    const float inv = 1.0f / red[row][0];
    for (int i = sub; i < SEQ; i += 8) r[i] *= inv;
  }
  __syncthreads();

  // ---- 3) async LDS -> HBM stream of the p_attn strip (no VGPR traffic;
  //         overlaps with the PV WMMA loop below). 16 B per lane per issue.
  {
    float* dst = out_p + ((size_t)bh * SEQ + m0) * SEQ;
    // Low 32 bits of a generic LDS pointer are the LDS byte offset
    // (aperture layout: addr[63:32]=shared_base, addr[31:0]=LDS address).
    unsigned lbase = (unsigned)(uintptr_t)sm + (unsigned)threadIdx.x * 16u;
    unsigned long long gbase =
        (unsigned long long)(uintptr_t)dst + (unsigned long long)threadIdx.x * 16u;
#pragma unroll 4
    for (int i = 0; i < 64; ++i) {                // 64 * 128thr * 16B = 128KB
      unsigned long long gaddr = gbase + (unsigned long long)i * (128u * 16u);
      unsigned laddr = lbase + (unsigned)i * (128u * 16u);
      asm volatile("global_store_async_from_lds_b128 %0, %1, off"
                   :: "v"(gaddr), "v"(laddr)
                   : "memory");
    }
  }

  // ---- 4) x = P @ V  (wave w owns output cols n0 = 16*w .. 16*w+15) ----
  {
    const int n0 = wave * 16;
    v8f acc = {};
    for (int j0 = 0; j0 < SEQ; j0 += 32) {
      v16h a = frag_nt_f32(sm + j0, SEQ);                           // P tile
      v16h b = fragB_kn_f16(vbase + (size_t)j0 * DK + n0, DK);      // V tile
      acc = wmma_f16(a, b, acc);
    }
    const int b_ = bh >> 4, h = bh & (HEADS - 1);
    float* xb = out_x + ((size_t)b_ * SEQ + m0) * DMODEL + h * DK + n0;
#pragma unroll
    for (int r = 0; r < 8; ++r)
      xb[(size_t)(r + 8 * g) * DMODEL + nl] = acc[r];
  }

  // ---- 5) make sure the async p_attn stores have retired ----
#if __has_builtin(__builtin_amdgcn_s_wait_asynccnt)
  __builtin_amdgcn_s_wait_asynccnt(0);
#else
  asm volatile("s_wait_asynccnt 0" ::: "memory");
#endif
}

// ---------------------------------------------------------------------------
// Host launcher
// ---------------------------------------------------------------------------
extern "C" void kernel_launch(void* const* d_in, const int* in_sizes, int n_in,
                              void* d_out, int out_size, void* d_ws,
                              size_t ws_size, hipStream_t stream) {
  (void)in_sizes; (void)n_in; (void)out_size; (void)ws_size;

  const float* q = (const float*)d_in[0];
  const float* k = (const float*)d_in[1];
  const float* v = (const float*)d_in[2];
  const float* w = (const float*)d_in[3];

  float* out_x = (float*)d_out;                                   // [B,S,D]
  float* out_p = out_x + (size_t)BATCH * SEQ * DMODEL;            // [B,H,S,S]

  const size_t head_elems = (size_t)BATCH * HEADS * SEQ * DK;     // 8.39M
  _Float16* qh = (_Float16*)d_ws;
  _Float16* kh = qh + head_elems;
  _Float16* vh = kh + head_elems;                                 // 50.3 MB ws

  // Projections: grid covers (N=1024/64, M=8192/64, qkv)
  proj_qkv<<<dim3(16, 128, 3), dim3(128), 0, stream>>>(q, k, v, w, qh, kh, vh);

  // Fused attention: one block per (16 query rows, b*h); 128KB dynamic LDS
  attn_fused<<<dim3(1, SEQ / 16, BATCH * HEADS), dim3(128),
               16 * SEQ * sizeof(float), stream>>>(qh, kh, vh, out_p, out_x);
}
